// MHA_3255585210404
// MI455X (gfx1250) — matrix operand order repaired and checked by reading, not compiled, not run on hardware
//
#include <hip/hip_runtime.h>

typedef __attribute__((ext_vector_type(16))) _Float16 v16h;
typedef __attribute__((ext_vector_type(8)))  _Float16 v8h;
typedef __attribute__((ext_vector_type(4)))  _Float16 v4h;
typedef __attribute__((ext_vector_type(2)))  _Float16 v2h;
typedef __attribute__((ext_vector_type(8)))  float    v8f;
typedef __attribute__((ext_vector_type(4)))  float    v4f;

#define D_MODEL 1024
#define SEQ     2048
#define BATCH   2
#define HEADS   16
#define HDIM    64
#define MTOT    (BATCH * SEQ)   // 4096

// ds_swizzle imm: group-of-32 mode, and_mask=31, or_mask=0, xor_mask=16
// -> lane pairs (i, i^16) exchange values (combine the two 16-lane halves)
#define SWZ_XOR16 0x401F

union ABFrag { v16h v; v8h h[2]; unsigned int u[8]; };
union PkH2   { v2h h2; unsigned int u; };

__device__ __forceinline__ v16h make_frag(const _Float16* p0, const _Float16* p1) {
  ABFrag f;
  f.h[0] = *(const v8h*)p0;
  f.h[1] = *(const v8h*)p1;
  return f.v;
}

__device__ __forceinline__ float swz16_f(float x) {
  return __int_as_float(__builtin_amdgcn_ds_swizzle(__float_as_int(x), SWZ_XOR16));
}

// ---------------- elementwise f32 -> f16 ----------------
__global__ void cvt_f32_to_f16(const float* __restrict__ in, _Float16* __restrict__ out, int n) {
  int i = (blockIdx.x * blockDim.x + threadIdx.x) * 4;
  if (i + 3 < n) {
    v4f x = *(const v4f*)(in + i);
    v4h y;
    y[0] = (_Float16)x[0]; y[1] = (_Float16)x[1];
    y[2] = (_Float16)x[2]; y[3] = (_Float16)x[3];
    *(v4h*)(out + i) = y;
  }
}

// ---------------- W [K,N] f32 -> Wt [N,K] f16 (transpose) ----------------
__global__ void cvt_w_t(const float* __restrict__ W, _Float16* __restrict__ Wt) {
  int t = blockIdx.x * blockDim.x + threadIdx.x;    // 0 .. 1024*1024-1
  int n = t >> 10, k = t & 1023;
  Wt[n * D_MODEL + k] = (_Float16)W[k * D_MODEL + n];
}

// ---------------- V [4096,1024] f16 -> Vt [b,h,d,s] f16 ----------------
__global__ void transpose_v(const _Float16* __restrict__ Vh, _Float16* __restrict__ Vt) {
  int i = blockIdx.x * blockDim.x + threadIdx.x;
  int srow = i >> 10;           // 0..4095
  int n = i & 1023;
  int b = srow >> 11, s = srow & 2047;
  int h = n >> 6, d = n & 63;
  Vt[(((b * HEADS + h) * HDIM + d) << 11) + s] = Vh[i];
}

// ---------------- GEMM: Y[M,N] = Xh[M,K] * Wt[N,K]^T + bias ----------------
// block = 256 threads = 8 waves; each wave computes a 32(M) x 64(N) strip.
// block tile = 128M x 128N ; waves arranged 4(M) x 2(N).
template <bool OUT_F16>
__global__ void gemm_wmma_kernel(const _Float16* __restrict__ X,
                                 const _Float16* __restrict__ Wt,
                                 const float* __restrict__ bias,
                                 void* __restrict__ out_) {
  const int K = D_MODEL, N = D_MODEL;
  int lane = threadIdx.x & 31;
  int wave = threadIdx.x >> 5;
  int g = lane >> 4, idx = lane & 15;
  int m0 = blockIdx.y * 128 + (wave & 3) * 32;
  int n0 = blockIdx.x * 128 + (wave >> 2) * 64;

  v8f acc[2][4] = {};
  const _Float16* xr0 = X + (m0 + idx) * K;        // A: lane row = idx
  const _Float16* xr1 = X + (m0 + 16 + idx) * K;
  for (int k0 = 0; k0 < K; k0 += 32) {
    // A layout: halves 0-7 -> K = k0+g*8.. ; halves 8-15 -> K = k0+16+g*8..
    v16h a0 = make_frag(xr0 + k0 + g * 8, xr0 + k0 + 16 + g * 8);
    v16h a1 = make_frag(xr1 + k0 + g * 8, xr1 + k0 + 16 + g * 8);
#pragma unroll
    for (int t = 0; t < 4; ++t) {
      // B layout: lane col = idx ; halves 0-7 -> K = k0+g*8.. ; halves 8-15 -> K = k0+16+g*8..
      const _Float16* br = Wt + (n0 + t * 16 + idx) * K + k0;
      v16h bfr = make_frag(br + g * 8, br + 16 + g * 8);
      acc[0][t] = __builtin_amdgcn_wmma_f32_16x16x32_f16(false, a0, false, bfr,
                                                         (short)0, acc[0][t], false, false);
      acc[1][t] = __builtin_amdgcn_wmma_f32_16x16x32_f16(false, a1, false, bfr,
                                                         (short)0, acc[1][t], false, false);
    }
  }
#pragma unroll
  for (int t = 0; t < 4; ++t) {
    int n = n0 + t * 16 + idx;
    float bv = bias[n];
#pragma unroll
    for (int r = 0; r < 8; ++r) {
      int m = m0 + r + g * 8;          // C layout: vgpr r -> row r + g*8
      float v0 = acc[0][t][r] + bv;
      float v1 = acc[1][t][r] + bv;
      if (OUT_F16) {
        ((_Float16*)out_)[m * N + n] = (_Float16)v0;
        ((_Float16*)out_)[(m + 16) * N + n] = (_Float16)v1;
      } else {
        ((float*)out_)[m * N + n] = v0;
        ((float*)out_)[(m + 16) * N + n] = v1;
      }
    }
  }
}

// ---------------- causal flash attention (transposed-score layout) ----------------
// One wave per 16-query tile. S^T = K*Q^T so each lane owns one query's scores:
// softmax state is a per-lane scalar; cross-half combine = 1 ds_swizzle xor16.
// O^T = V^T * P^T accumulated in 4 C-tiles (d rows, query cols).
__global__ void attn_wmma_kernel(const _Float16* __restrict__ Qh,
                                 const _Float16* __restrict__ Kh,
                                 const _Float16* __restrict__ Vt,
                                 _Float16* __restrict__ ctx) {
  int lane = threadIdx.x & 31;
  int wave = threadIdx.x >> 5;
  int g = lane >> 4, idx = lane & 15;
  int qt = blockIdx.x * 8 + wave;
  int q0 = qt * 16;
  int h = blockIdx.y, b = blockIdx.z;
  const float scale = 0.125f;   // 1/sqrt(64)

  // Q as B-operand (col = query q0+idx ; halves 0-7 d = g*8.., halves 8-15 d = 16+g*8.. (+dbase))
  const _Float16* qrow = Qh + (b * SEQ + q0 + idx) * D_MODEL + h * HDIM;
  v16h bq0 = make_frag(qrow + g * 8,      qrow + 16 + g * 8);        // d 0..31
  v16h bq1 = make_frag(qrow + 32 + g * 8, qrow + 48 + g * 8);        // d 32..63

  float rmax = -1e30f, rsum = 0.0f;
  v8f o[4] = {};   // O^T tiles: row d = t*16 + r + g*8 ; col q = idx
  const _Float16* vbase = Vt + (size_t)((b * HEADS + h) * HDIM) * SEQ;

  int nkb = qt / 2 + 1;    // # of 32-key blocks (causal)
  for (int kb = 0; kb < nkb; ++kb) {
    int key0 = kb * 32;
    // ---- S^T tiles: t0 = keys key0..+15, t1 = keys key0+16..+31 (rows), queries (cols)
    v8f t0 = {}, t1 = {};
    {
      const _Float16* k0r = Kh + (b * SEQ + key0 + idx) * D_MODEL + h * HDIM;  // A rows = keys
      const _Float16* k1r = k0r + 16 * D_MODEL;
      v16h ak;
      ak = make_frag(k0r + g * 8,      k0r + 16 + g * 8);   // d 0..31
      t0 = __builtin_amdgcn_wmma_f32_16x16x32_f16(false, ak, false, bq0, (short)0, t0, false, false);
      ak = make_frag(k0r + 32 + g * 8, k0r + 48 + g * 8);   // d 32..63
      t0 = __builtin_amdgcn_wmma_f32_16x16x32_f16(false, ak, false, bq1, (short)0, t0, false, false);
      ak = make_frag(k1r + g * 8,      k1r + 16 + g * 8);
      t1 = __builtin_amdgcn_wmma_f32_16x16x32_f16(false, ak, false, bq0, (short)0, t1, false, false);
      ak = make_frag(k1r + 32 + g * 8, k1r + 48 + g * 8);
      t1 = __builtin_amdgcn_wmma_f32_16x16x32_f16(false, ak, false, bq1, (short)0, t1, false, false);
    }
    // ---- scale + causal mask. element: key = key0 + tile*16 + r + g*8 ; q = q0 + idx
    bool last = (kb == nkb - 1);
    int q = q0 + idx;
    float sc[16];
#pragma unroll
    for (int r = 0; r < 8; ++r) {
      float a0 = t0[r] * scale;
      float a1 = t1[r] * scale;
      if (last) {
        int k0i = key0 + r + g * 8;
        if (k0i > q)      a0 = -1e30f;
        if (k0i + 16 > q) a1 = -1e30f;
      }
      sc[r] = a0; sc[8 + r] = a1;
    }
    // ---- online softmax: per-lane (per-query) scalar state
    float mx = sc[0];
#pragma unroll
    for (int i = 1; i < 16; ++i) mx = fmaxf(mx, sc[i]);
    mx = fmaxf(mx, swz16_f(mx));           // combine the two half-wave partials
    float nm = fmaxf(rmax, mx);
    float corr = __expf(rmax - nm);
    float ps = 0.0f;
#pragma unroll
    for (int i = 0; i < 16; ++i) { sc[i] = __expf(sc[i] - nm); ps += sc[i]; }
    ps += swz16_f(ps);
    rsum = rsum * corr + ps;
    rmax = nm;
#pragma unroll
    for (int t = 0; t < 4; ++t)
#pragma unroll
      for (int r = 0; r < 8; ++r) o[t][r] *= corr;

    // ---- build P^T B-fragment (32 keys x 16 queries) from registers.
    // Lane already holds (col q): keys key0 + (g*8..g*8+7) and key0+16+(g*8..+7).
    // B needs halves h -> key = key0 + h + g*16: own pairs + cross-half swap.
    unsigned int pk0[4], pk1[4], sw0[4], sw1[4];
#pragma unroll
    for (int j = 0; j < 4; ++j) {
      PkH2 c0, c1;
      c0.h2[0] = (_Float16)sc[2 * j];     c0.h2[1] = (_Float16)sc[2 * j + 1];
      c1.h2[0] = (_Float16)sc[8 + 2 * j]; c1.h2[1] = (_Float16)sc[8 + 2 * j + 1];
      pk0[j] = c0.u;
      pk1[j] = c1.u;
      sw0[j] = (unsigned int)__builtin_amdgcn_ds_swizzle((int)pk0[j], SWZ_XOR16);
      sw1[j] = (unsigned int)__builtin_amdgcn_ds_swizzle((int)pk1[j], SWZ_XOR16);
    }
    ABFrag bp;
#pragma unroll
    for (int j = 0; j < 4; ++j) {
      bp.u[j]     = g ? sw1[j] : pk0[j];   // halves 0-7:  keys g*16 + 0..7
      bp.u[4 + j] = g ? pk1[j] : sw0[j];   // halves 8-15: keys g*16 + 8..15
    }
    // ---- O^T += V^T x P^T
#pragma unroll
    for (int t = 0; t < 4; ++t) {
      const _Float16* vr = vbase + (size_t)(t * 16 + idx) * SEQ + key0;  // A rows = d
      v16h av = make_frag(vr + g * 16, vr + g * 16 + 8);
      o[t] = __builtin_amdgcn_wmma_f32_16x16x32_f16(false, av, false, bp.v, (short)0, o[t], false, false);
    }
  }
  // ---- normalize and store: lane owns ctx row (b, q0+idx); d contiguous per tile
  float inv = 1.0f / rsum;
  _Float16* crow = ctx + (size_t)(b * SEQ + q0 + idx) * D_MODEL + h * HDIM;
#pragma unroll
  for (int t = 0; t < 4; ++t) {
    v8h outv;
#pragma unroll
    for (int r = 0; r < 8; ++r) outv[r] = (_Float16)(o[t][r] * inv);
    *(v8h*)(crow + t * 16 + g * 8) = outv;   // 16B store, d = t*16 + g*8 .. +7
  }
}

// ---------------- host launch ----------------
extern "C" void kernel_launch(void* const* d_in, const int* in_sizes, int n_in,
                              void* d_out, int out_size, void* d_ws, size_t ws_size,
                              hipStream_t stream) {
  (void)in_sizes; (void)n_in; (void)out_size; (void)ws_size;
  const float* x  = (const float*)d_in[0];
  const float* Wq = (const float*)d_in[1];
  const float* bq = (const float*)d_in[2];
  const float* Wk = (const float*)d_in[3];
  const float* bk = (const float*)d_in[4];
  const float* Wv = (const float*)d_in[5];
  const float* bv = (const float*)d_in[6];
  const float* Wo = (const float*)d_in[7];
  const float* bo = (const float*)d_in[8];

  char* ws = (char*)d_ws;
  const size_t MB = 1ull << 20;
  _Float16* xh  = (_Float16*)(ws + 0 * MB);    // 8 MB
  _Float16* wqt = (_Float16*)(ws + 8 * MB);    // 2 MB each
  _Float16* wkt = (_Float16*)(ws + 10 * MB);
  _Float16* wvt = (_Float16*)(ws + 12 * MB);
  _Float16* wot = (_Float16*)(ws + 14 * MB);
  _Float16* qh  = (_Float16*)(ws + 16 * MB);   // 8 MB each
  _Float16* kh  = (_Float16*)(ws + 24 * MB);
  _Float16* vh  = (_Float16*)(ws + 32 * MB);
  _Float16* vt  = (_Float16*)(ws + 40 * MB);
  _Float16* ctx = (_Float16*)(ws + 48 * MB);

  const int nX = MTOT * D_MODEL;              // 4096*1024
  cvt_f32_to_f16<<<nX / (256 * 4), 256, 0, stream>>>(x, xh, nX);
  cvt_w_t<<<(D_MODEL * D_MODEL) / 256, 256, 0, stream>>>(Wq, wqt);
  cvt_w_t<<<(D_MODEL * D_MODEL) / 256, 256, 0, stream>>>(Wk, wkt);
  cvt_w_t<<<(D_MODEL * D_MODEL) / 256, 256, 0, stream>>>(Wv, wvt);
  cvt_w_t<<<(D_MODEL * D_MODEL) / 256, 256, 0, stream>>>(Wo, wot);

  dim3 ggrid(D_MODEL / 128, MTOT / 128);
  gemm_wmma_kernel<true><<<ggrid, 256, 0, stream>>>(xh, wqt, bq, qh);
  gemm_wmma_kernel<true><<<ggrid, 256, 0, stream>>>(xh, wkt, bk, kh);
  gemm_wmma_kernel<true><<<ggrid, 256, 0, stream>>>(xh, wvt, bv, vh);

  transpose_v<<<nX / 256, 256, 0, stream>>>(vh, vt);

  attn_wmma_kernel<<<dim3(SEQ / 16 / 8, HEADS, BATCH), 256, 0, stream>>>(qh, kh, vt, ctx);

  gemm_wmma_kernel<false><<<ggrid, 256, 0, stream>>>(ctx, wot, bo, (float*)d_out);
}
